// Dilateformer3_47983374631384
// MI455X (gfx1250) — matrix operand order repaired and checked by reading, not compiled, not run on hardware
//
#include <hip/hip_runtime.h>

// ---------------------------------------------------------------------------
// Dilateformer block for gfx1250 (MI455X): wave32, WMMA f16, LDS tiling with
// TDM (tensor_load_to_lds) panel staging.
//   x:(8,56,56,288) f32 -> qkv GEMM (f16 WMMA, TDM-staged LDS tiles) ->
//   dilated 3x3 attention (3 dilations x 2 heads x head_dim 48, fp32 VALU) ->
//   proj GEMM (f16 WMMA, TDM-staged LDS tiles)
// ---------------------------------------------------------------------------

typedef __attribute__((ext_vector_type(16))) _Float16 v16h;
typedef __attribute__((ext_vector_type(8)))  _Float16 v8h;
typedef __attribute__((ext_vector_type(4)))  _Float16 v4h;
typedef __attribute__((ext_vector_type(8)))  float    v8f;
typedef __attribute__((ext_vector_type(4)))  float    v4f;
typedef __attribute__((ext_vector_type(4)))  unsigned int u32x4;
typedef __attribute__((ext_vector_type(8)))  int      i32x8;
typedef __attribute__((ext_vector_type(4)))  int      i32x4;

#define CDIM   288
#define QKVDIM 864
#define BATCH  8
#define HIMG   56
#define WIMG   56
#define HWPIX  (HIMG * WIMG)          // 3136
#define NPIX   (BATCH * HWPIX)        // 25088
#define HEADD  48

#define KDIM   288                    // GEMM K (both GEMMs)
#define BM     128                    // block M tile
#define RS0    264                    // LDS row stride, main chunk (halves; 528B)
#define RS1    40                     // LDS row stride, tail chunk (halves; 80B)

union V16 { v16h v; v8h h[2]; };

// ---------------------------------------------------------------------------
// fp32 -> f16 conversion (vectorized 4-wide; all element counts are %4 == 0)
// ---------------------------------------------------------------------------
__global__ void cvt_f32_to_f16(const float* __restrict__ src,
                               _Float16* __restrict__ dst, int n4) {
  int i = blockIdx.x * blockDim.x + threadIdx.x;
  if (i >= n4) return;
  v4f x = ((const v4f*)src)[i];
  v4h h;
  h[0] = (_Float16)x[0]; h[1] = (_Float16)x[1];
  h[2] = (_Float16)x[2]; h[3] = (_Float16)x[3];
  ((v4h*)dst)[i] = h;
}

// ---------------------------------------------------------------------------
// TDM: one 2D tile load (rows x tile_d0 f16 elements) from a row-major tensor
// with row stride `stride` (elements) into LDS at byte offset lds_off, with
// (pad_amount_code+1)*... : pad 4 dwords (16B) appended after every row
// (row byte length must be 2^(pad_interval_code+3)).
// D# field packing per CDNA5 ISA.
// ---------------------------------------------------------------------------
__device__ __forceinline__ void tdm_load_2d(unsigned lds_off,
                                            const _Float16* gptr,
                                            int tile_d0, int rows, int stride,
                                            unsigned pad_interval_code) {
  unsigned long long ga = (unsigned long long)(size_t)gptr;
  u32x4 g0;
  g0[0] = 1u;                                    // count=1 (valid user desc)
  g0[1] = lds_off;                               // lds_addr (bytes)
  g0[2] = (unsigned)(ga & 0xffffffffu);          // global_addr[31:0]
  g0[3] = (unsigned)((ga >> 32) & 0x01ffffffu)   // global_addr[56:32]
        | 0x80000000u;                           // type=2 ("image")
  i32x8 g1;
  g1[0] = (int)((1u << 16)                       // data_size = 2 bytes
              | (1u << 20)                       // pad_enable
              | (pad_interval_code << 22)        // pad after 2^(c+3) bytes
              | (3u << 25));                     // pad_amount = 4 dwords (16B)
  g1[1] = (int)(((unsigned)stride & 0xffffu) << 16);      // tensor_dim0 lo16
  g1[2] = (int)(0xffff0000u);                    // dim0 hi=0, tensor_dim1 lo16
  g1[3] = (int)((0x7fffu) | ((unsigned)tile_d0 << 16));   // dim1 hi, tile_dim0
  g1[4] = (int)((unsigned)rows & 0xffffu);       // tile_dim1; tile_dim2=0
  g1[5] = stride;                                // tensor_dim0_stride lo32
  g1[6] = 0;                                     // stride hi; dim1_stride lo
  g1[7] = 0;
  i32x4 gz = { 0, 0, 0, 0 };                     // 2D tile: groups 2/3 unused
#if defined(__clang_major__) && __clang_major__ >= 23
  i32x8 gz8 = { 0, 0, 0, 0, 0, 0, 0, 0 };
  __builtin_amdgcn_tensor_load_to_lds(g0, g1, gz, gz, gz8, 0);
#else
  __builtin_amdgcn_tensor_load_to_lds(g0, g1, gz, gz, 0);
#endif
}

// ---------------------------------------------------------------------------
// LDS block-tiled GEMM: out[m,n] = sum_k A[m,k] * Wt[n,k] + bias[n]
//   A:  row-major [M, 288] f16.  Wt: row-major [N, 288] f16 (K contiguous).
//   Block = 256 threads (8 waves). Block tile BM=128 x BN=NT*16.
//   Panels staged into LDS by 4 TDM descriptors issued by wave 0:
//     main chunk: k[0,256)  rows of 512B, padded +16B -> stride 528B (RS0)
//     tail chunk: k[256,288) rows of 64B, padded +16B -> stride  80B (RS1)
//   Both strides are bank-conflict-free for b128 fragment loads
//   (132 dw == 4 mod 64 tiles all 64 banks; 20 dw cycles 16 distinct starts).
//   Wave w owns rows w*16..w*16+15, computes NT 16x16 tiles per K-step,
//   reusing its A fragment NT times. 8 main + 1 tail K-steps of
//   v_wmma_f32_16x16x32_f16. Fragment layouts per CDNA5 ISA 7.12.2:
//     A 16x32: lanes 0-15 row m0+L, k {0..7,16..23}; lanes 16-31 k {8..15,24..31}
//     B 32x16: lanes 0-15 col n0+L, k {0..7,16..23}; lanes 16-31 k {8..15,24..31}
//     C/D:     lane = col (mod 16), VGPR r = row r (+8 for lanes 16-31)
//   Bias pre-splatted into accumulators. grid = (N/(NT*16), M/128).
// ---------------------------------------------------------------------------
template <int NT, bool HALF_OUT>
__global__ __launch_bounds__(256) void gemm_wmma_lds(
    const _Float16* __restrict__ A, const _Float16* __restrict__ Wt,
    const float* __restrict__ bias, void* __restrict__ out, int N) {
  // LDS region offsets (halves). Single static LDS object => base offset 0.
  constexpr int A0H = 0;
  constexpr int A1H = A0H + BM * RS0;        // 33792
  constexpr int B0H = A1H + BM * RS1;        // 38912
  constexpr int B1H = B0H + NT * 16 * RS0;
  constexpr int TOTH = B1H + NT * 16 * RS1;  // NT=6: 68096 halves = 136.2 KB
  __shared__ _Float16 smem[TOTH];

  const int tid = threadIdx.x;

  // ---- stage A (BM x 288) and B (NT*16 x 288) panels via TDM ----
  if (tid < 32) {
    const _Float16* Ag = A + (size_t)(blockIdx.y * BM) * KDIM;
    const _Float16* Bg = Wt + (size_t)(blockIdx.x * (NT * 16)) * KDIM;
    tdm_load_2d((unsigned)(A0H * 2), Ag,        256, BM,      KDIM, 6u);
    tdm_load_2d((unsigned)(A1H * 2), Ag + 256,  32,  BM,      KDIM, 3u);
    tdm_load_2d((unsigned)(B0H * 2), Bg,        256, NT * 16, KDIM, 6u);
    tdm_load_2d((unsigned)(B1H * 2), Bg + 256,  32,  NT * 16, KDIM, 3u);
    __builtin_amdgcn_s_wait_tensorcnt(0);
  }
  __syncthreads();

  // ---- per-wave compute: 16 rows x NT*16 cols ----
  const int wv    = tid >> 5;
  const int lane  = tid & 31;
  const int lcol  = lane & 15;
  const int lhalf = lane >> 4;

  v8f c[NT];
#pragma unroll
  for (int t = 0; t < NT; ++t) {
    const float bv = bias[blockIdx.x * (NT * 16) + t * 16 + lcol];
    c[t] = (v8f){ bv, bv, bv, bv, bv, bv, bv, bv };
  }

  const _Float16* a0 = &smem[A0H + (wv * 16 + lcol) * RS0 + (lhalf ? 8 : 0)];
  const _Float16* b0 = &smem[B0H + lcol * RS0 + (lhalf ? 8 : 0)];
  const _Float16* a1 = &smem[A1H + (wv * 16 + lcol) * RS1 + (lhalf ? 8 : 0)];
  const _Float16* b1 = &smem[B1H + lcol * RS1 + (lhalf ? 8 : 0)];

#pragma unroll
  for (int ks = 0; ks < 8; ++ks) {          // main chunk: k = ks*32 .. +31
    const int k0 = ks * 32;
    V16 ua;
    ua.h[0] = *(const v8h*)(a0 + k0);       // ds_load_b128
    ua.h[1] = *(const v8h*)(a0 + k0 + 16);
#pragma unroll
    for (int t = 0; t < NT; ++t) {
      V16 ub;
      const _Float16* bp = b0 + t * 16 * RS0 + k0;
      ub.h[0] = *(const v8h*)bp;
      ub.h[1] = *(const v8h*)(bp + 16);
      c[t] = __builtin_amdgcn_wmma_f32_16x16x32_f16(
          false, ua.v, false, ub.v, (short)0, c[t], false, false);
    }
  }
  {                                         // tail chunk: k = 256 .. 287
    V16 ua;
    ua.h[0] = *(const v8h*)(a1);
    ua.h[1] = *(const v8h*)(a1 + 16);
#pragma unroll
    for (int t = 0; t < NT; ++t) {
      V16 ub;
      const _Float16* bp = b1 + t * 16 * RS1;
      ub.h[0] = *(const v8h*)bp;
      ub.h[1] = *(const v8h*)(bp + 16);
      c[t] = __builtin_amdgcn_wmma_f32_16x16x32_f16(
          false, ua.v, false, ub.v, (short)0, c[t], false, false);
    }
  }

  // ---- epilogue ----
  const int rowg = blockIdx.y * BM + wv * 16 + lhalf * 8;
#pragma unroll
  for (int t = 0; t < NT; ++t) {
    const int colg = blockIdx.x * (NT * 16) + t * 16 + lcol;
    if (HALF_OUT) {
      _Float16* O = (_Float16*)out;
#pragma unroll
      for (int r = 0; r < 8; ++r)
        O[(size_t)(rowg + r) * N + colg] = (_Float16)c[t][r];
    } else {
      float* O = (float*)out;
#pragma unroll
      for (int r = 0; r < 8; ++r)
        O[(size_t)(rowg + r) * N + colg] = c[t][r];
    }
  }
}

// ---------------------------------------------------------------------------
// Dilated 3x3 attention. One thread per (pixel, dilation-group g, head j).
// qkv layout: [pix, 864] f16 with channel order [qkv(3), dil(3), head(2), 48].
// Zero-padded unfold semantics: OOB neighbor => logit 0 (participates in
// softmax) and zero V contribution. Output y channel order [dil, head, 48].
// ---------------------------------------------------------------------------
__global__ void dilate_attn(const _Float16* __restrict__ qkv,
                            _Float16* __restrict__ y) {
  const int t   = blockIdx.x * blockDim.x + threadIdx.x;   // 25088*6 threads
  const int pix = t / 6;
  const int r   = t - pix * 6;
  const int g   = r >> 1;          // dilation group: dil = g+1
  const int j   = r & 1;           // head within group
  const int b   = pix / HWPIX;
  const int p   = pix - b * HWPIX;
  const int py  = p / WIMG;
  const int px  = p - py * WIMG;
  const int dil = g + 1;
  const int coff = g * 96 + j * HEADD;

  const _Float16* qb = qkv + (size_t)pix * QKVDIM + coff;
  v8h qh[6];
#pragma unroll
  for (int i = 0; i < 6; ++i) qh[i] = ((const v8h*)qb)[i];

  const float scale = 0.14433756729740643f;  // 48^-0.5
  float s[9];
#pragma unroll
  for (int n = 0; n < 9; ++n) {
    const int ny = py + (n / 3 - 1) * dil;
    const int nx = px + (n % 3 - 1) * dil;
    float acc = 0.f;
    if ((unsigned)ny < (unsigned)HIMG && (unsigned)nx < (unsigned)WIMG) {
      const int npix = b * HWPIX + ny * WIMG + nx;
      const _Float16* kb = qkv + (size_t)npix * QKVDIM + CDIM + coff;
#pragma unroll
      for (int i = 0; i < 6; ++i) {
        v8h kh = ((const v8h*)kb)[i];
#pragma unroll
        for (int e = 0; e < 8; ++e)
          acc += (float)qh[i][e] * (float)kh[e];   // v_fma_mix_f32 path
      }
    }
    s[n] = acc * scale;
  }

  float m = s[0];
#pragma unroll
  for (int n = 1; n < 9; ++n) m = fmaxf(m, s[n]);
  float sum = 0.f;
#pragma unroll
  for (int n = 0; n < 9; ++n) { s[n] = __expf(s[n] - m); sum += s[n]; }
  const float inv = 1.f / sum;

  float acc[HEADD];
#pragma unroll
  for (int i = 0; i < HEADD; ++i) acc[i] = 0.f;
#pragma unroll
  for (int n = 0; n < 9; ++n) {
    const int ny = py + (n / 3 - 1) * dil;
    const int nx = px + (n % 3 - 1) * dil;
    if ((unsigned)ny < (unsigned)HIMG && (unsigned)nx < (unsigned)WIMG) {
      const float pw = s[n] * inv;
      const int npix = b * HWPIX + ny * WIMG + nx;
      const _Float16* vb = qkv + (size_t)npix * QKVDIM + 2 * CDIM + coff;
#pragma unroll
      for (int i = 0; i < 6; ++i) {
        v8h vh = ((const v8h*)vb)[i];
#pragma unroll
        for (int e = 0; e < 8; ++e)
          acc[i * 8 + e] += pw * (float)vh[e];
      }
    }
  }

  _Float16* yb = y + (size_t)pix * CDIM + coff;
#pragma unroll
  for (int i = 0; i < 6; ++i) {
    v8h o;
#pragma unroll
    for (int e = 0; e < 8; ++e) o[e] = (_Float16)acc[i * 8 + e];
    ((v8h*)yb)[i] = o;
  }
}

// ---------------------------------------------------------------------------
// Host side
// ---------------------------------------------------------------------------
extern "C" void kernel_launch(void* const* d_in, const int* in_sizes, int n_in,
                              void* d_out, int out_size, void* d_ws, size_t ws_size,
                              hipStream_t stream) {
  const float* x      = (const float*)d_in[0];   // (8,56,56,288)
  const float* w_qkv  = (const float*)d_in[1];   // (864,288)
  const float* b_qkv  = (const float*)d_in[2];   // (864)
  const float* w_proj = (const float*)d_in[3];   // (288,288)
  const float* b_proj = (const float*)d_in[4];   // (288)
  float* out = (float*)d_out;                    // (8,56,56,288) f32

  char* ws = (char*)d_ws;
  size_t off = 0;
  auto take = [&](size_t bytes) {
    void* pp = ws + off;
    off = (off + bytes + 255) & ~(size_t)255;
    return pp;
  };
  _Float16* xh    = (_Float16*)take((size_t)NPIX * CDIM * 2);    // 14.5 MB
  _Float16* wqh   = (_Float16*)take((size_t)QKVDIM * CDIM * 2);  // 0.5 MB
  _Float16* wph   = (_Float16*)take((size_t)CDIM * CDIM * 2);    // 0.17 MB
  _Float16* qkvh  = (_Float16*)take((size_t)NPIX * QKVDIM * 2);  // 43.4 MB
  _Float16* yh    = (_Float16*)take((size_t)NPIX * CDIM * 2);    // 14.5 MB
  (void)ws_size; (void)in_sizes; (void)n_in; (void)out_size;

  // 1) converts
  {
    int n4 = NPIX * CDIM / 4;
    cvt_f32_to_f16<<<(n4 + 255) / 256, 256, 0, stream>>>(x, xh, n4);
    n4 = QKVDIM * CDIM / 4;
    cvt_f32_to_f16<<<(n4 + 255) / 256, 256, 0, stream>>>(w_qkv, wqh, n4);
    n4 = CDIM * CDIM / 4;
    cvt_f32_to_f16<<<(n4 + 255) / 256, 256, 0, stream>>>(w_proj, wph, n4);
  }

  // 2) qkv GEMM: M=25088, N=864 -> f16 qkv, fused bias.
  //    Block tile 128x96 (NT=6), grid (864/96, 25088/128) = (9, 196).
  {
    dim3 grid(QKVDIM / 96, NPIX / BM);
    gemm_wmma_lds<6, true><<<grid, 256, 0, stream>>>(xh, wqh, b_qkv,
                                                     (void*)qkvh, QKVDIM);
  }

  // 3) dilated attention: one thread per (pixel, dil, head)
  {
    int nthr = NPIX * 6;                          // 150528, %256 == 0
    dilate_attn<<<nthr / 256, 256, 0, stream>>>(qkvh, yh);
  }

  // 4) proj GEMM: M=25088, N=288 -> fp32 out, fused bias.
  //    Block tile 128x48 (NT=3), grid (288/48, 25088/128) = (6, 196).
  {
    dim3 grid(CDIM / 48, NPIX / BM);
    gemm_wmma_lds<3, false><<<grid, 256, 0, stream>>>(yh, wph, b_proj,
                                                      (void*)out, CDIM);
  }
}
